// BandSplitModule_15753940041762
// MI455X (gfx1250) — matrix operand order repaired and checked by reading, not compiled, not run on hardware
//
#include <hip/hip_runtime.h>
#include <hip/hip_bf16.h>

// ---------------------------------------------------------------------------
// BandSplit for MI455X (gfx1250): LDS-staged LayerNorm + bf16 WMMA GEMM.
// Output (8,128,36,2048) f32 written coalesced via an LDS transpose tile.
// ---------------------------------------------------------------------------

typedef __bf16 bf16_t;
typedef __attribute__((ext_vector_type(8)))  __bf16 v8bf;
typedef __attribute__((ext_vector_type(16))) __bf16 v16bf;
typedef __attribute__((ext_vector_type(8)))  float  v8f;

#define N_BANDS     36
#define BATCH       8
#define NFRAMES     2048
#define NBINS_TOT   1025
#define EMBED       128
#define MTILE       128      // frames per workgroup tile
#define DPAD_MAX    160      // max d (130) padded to multiple of 32
#define SXPAD       2        // f32 staging row pad (bank-conflict relief)
#define YSTRIDE     132      // output transpose tile stride (floats)

struct BandPtrs {
    const float* nw[N_BANDS];   // norm weights
    const float* nbb[N_BANDS];  // norm biases
    const float* lw[N_BANDS];   // linear weights (EMBED x d, row-major)
    const float* lb[N_BANDS];   // linear biases (EMBED)
};

__device__ __forceinline__ int band_nb(int band) {
    return band < 20 ? 16 : band < 30 ? 32 : band < 35 ? 64 : 65;
}
__device__ __forceinline__ int band_start(int band) {
    return band < 20 ? band * 16
         : band < 30 ? 320 + (band - 20) * 32
         : band < 35 ? 640 + (band - 30) * 64
         : 960;
}

__launch_bounds__(256, 1)
__global__ void bandsplit_kernel(const float* __restrict__ spec,
                                 float* __restrict__ out,
                                 BandPtrs P)
{
    extern __shared__ char smem[];

    const int tid  = threadIdx.x;
    const int bx   = blockIdx.x;
    const int band = bx >> 7;        // 128 blocks per band (8 batch * 16 t-tiles)
    const int rem  = bx & 127;
    const int b    = rem >> 4;
    const int t0   = (rem & 15) * MTILE;

    const int nb    = band_nb(band);
    const int start = band_start(band);
    const int d     = 2 * nb;
    const int dpad  = (d + 31) & ~31;    // 32 / 64 / 128 / 160
    const int SX    = dpad + SXPAD;

    // LDS layout (sized for worst case dpad = 160):
    float*  xs   = (float*)smem;                                      // MTILE*(DPAD_MAX+SXPAD) f32
    bf16_t* xn   = (bf16_t*)(smem + MTILE * (DPAD_MAX + SXPAD) * 4);  // MTILE*DPAD_MAX bf16
    bf16_t* wt2  = (bf16_t*)((char*)xn + MTILE * DPAD_MAX * 2);       // EMBED*DPAD_MAX bf16
    float*  swn  = (float*)((char*)wt2 + EMBED * DPAD_MAX * 2);       // DPAD_MAX f32
    float*  swb  = swn + DPAD_MAX;                                    // DPAD_MAX f32
    float*  ytile = xs;  // reused after LN phase: EMBED x MTILE, stride YSTRIDE

    // ---- stage layernorm params ----
    for (int j = tid; j < d; j += 256) {
        swn[j] = P.nw[band][j];
        swb[j] = P.nbb[band][j];
    }

    // ---- stage raw X tile: spec[b, start+j, t0+rr, 0:2] -> xs[rr][2j..2j+1]
    {
        const long specBase = (long)(b * NBINS_TOT + start) * NFRAMES * 2;
        const int total = nb * MTILE;
        for (int idx = tid; idx < total; idx += 256) {
            const int j  = idx >> 7;      // bin within band
            const int rr = idx & 127;     // frame within tile
            const float2 v =
                *(const float2*)(spec + specBase + ((long)j * NFRAMES + (t0 + rr)) * 2);
            xs[rr * SX + 2 * j]     = v.x;
            xs[rr * SX + 2 * j + 1] = v.y;
        }
    }

    // ---- stage weights as bf16, row-major (k contiguous), zero-pad k to dpad
    {
        const float* W = P.lw[band];
        const int tw = EMBED * d;
        for (int idx = tid; idx < tw; idx += 256) {
            const int e = idx / d;
            const int j = idx - e * d;
            wt2[e * dpad + j] = (bf16_t)W[idx];
        }
        const int pad = dpad - d;
        if (pad) {
            const int tp = EMBED * pad;
            for (int idx = tid; idx < tp; idx += 256) {
                const int e = idx / pad;
                const int j = d + (idx - e * pad);
                wt2[e * dpad + j] = (bf16_t)0.0f;
            }
        }
    }
    __syncthreads();

    // ---- LayerNorm: 2 threads per row, f32 stats, write bf16 panel ----
    {
        const int r = tid >> 1, h = tid & 1;
        const int half = dpad >> 1;
        const int j0 = h * half;
        const int j1 = min(j0 + half, d);
        float s = 0.f, ss = 0.f;
        for (int j = j0; j < j1; ++j) {
            const float v = xs[r * SX + j];
            s += v; ss += v * v;
        }
        s  += __shfl_xor(s, 1, 32);    // partner lane (wave32)
        ss += __shfl_xor(ss, 1, 32);
        const float mu   = s / (float)d;
        const float var  = ss / (float)d - mu * mu;
        const float rstd = rsqrtf(var + 1e-5f);
        for (int j = j0; j < j1; ++j) {
            const float v = (xs[r * SX + j] - mu) * rstd * swn[j] + swb[j];
            xn[r * dpad + j] = (bf16_t)v;
        }
        for (int j = j1; j < j0 + half; ++j)
            xn[r * dpad + j] = (bf16_t)0.0f;   // k padding
    }
    __syncthreads();

    // ---- bf16 WMMA GEMM: wave w owns rows [w*16, w*16+16) x all 128 cols ----
    {
        const int wave  = tid >> 5;
        const int lane  = tid & 31;
        const int m0    = wave * 16;
        const int l15   = lane & 15;
        const int m     = m0 + l15;
        const int koffA = (lane < 16) ? 0 : 8;    // A: per-lane K runs {0..7,16..23} / {8..15,24..31}
        const int koffB = (lane < 16) ? 0 : 8;    // B: per-lane K runs {0..7,16..23} / {8..15,24..31}
        const float* bias = P.lb[band];

        v8f acc[8];
        #pragma unroll
        for (int tn = 0; tn < 8; ++tn) {
            const float bv = bias[tn * 16 + l15];
            acc[tn] = (v8f){bv, bv, bv, bv, bv, bv, bv, bv};
        }

        const int nk = dpad >> 5;
        for (int ks = 0; ks < nk; ++ks) {
            const int k0 = ks * 32;
            union { v16bf v; v8bf h[2]; } A;
            A.h[0] = *(const v8bf*)(xn + m * dpad + k0 + koffA);
            A.h[1] = *(const v8bf*)(xn + m * dpad + k0 + 16 + koffA);
            #pragma unroll
            for (int tn = 0; tn < 8; ++tn) {
                const int n = tn * 16 + l15;
                union { v16bf v; v8bf h[2]; } Bm;
                Bm.h[0] = *(const v8bf*)(wt2 + n * dpad + k0 + koffB);
                Bm.h[1] = *(const v8bf*)(wt2 + n * dpad + k0 + koffB + 16);
                acc[tn] = __builtin_amdgcn_wmma_f32_16x16x32_bf16(
                    false, A.v, false, Bm.v, (short)0, acc[tn], false, false);
            }
        }

        // scatter accumulators into LDS transpose tile ytile[e][t]
        #pragma unroll
        for (int tn = 0; tn < 8; ++tn) {
            const int e = tn * 16 + l15;
            #pragma unroll
            for (int g = 0; g < 8; ++g) {
                const int mloc = (lane < 16) ? g : 8 + g;
                ytile[e * YSTRIDE + (m0 + mloc)] = acc[tn][g];
            }
        }
    }
    __syncthreads();

    // ---- coalesced output: out[((b*128+e)*36 + band)*2048 + t0 + tl] ----
    for (int idx = tid; idx < EMBED * MTILE; idx += 256) {
        const int e  = idx >> 7;
        const int tl = idx & 127;
        out[((long)(b * EMBED + e) * N_BANDS + band) * NFRAMES + t0 + tl] =
            ytile[e * YSTRIDE + tl];
    }
}

extern "C" void kernel_launch(void* const* d_in, const int* in_sizes, int n_in,
                              void* d_out, int out_size, void* d_ws, size_t ws_size,
                              hipStream_t stream) {
    (void)in_sizes; (void)n_in; (void)out_size; (void)d_ws; (void)ws_size;

    const float* spec = (const float*)d_in[0];
    BandPtrs P;
    for (int i = 0; i < N_BANDS; ++i) {
        P.nw[i]  = (const float*)d_in[1 + i];
        P.nbb[i] = (const float*)d_in[1 + N_BANDS + i];
        P.lw[i]  = (const float*)d_in[1 + 2 * N_BANDS + i];
        P.lb[i]  = (const float*)d_in[1 + 3 * N_BANDS + i];
    }

    const size_t smem = (size_t)MTILE * (DPAD_MAX + SXPAD) * 4   // f32 staging / y-tile
                      + (size_t)MTILE * DPAD_MAX * 2             // bf16 A panel
                      + (size_t)EMBED * DPAD_MAX * 2             // bf16 W panel
                      + (size_t)2 * DPAD_MAX * 4;                // norm params

    dim3 grid(N_BANDS * 128);   // 36 bands * 8 batch * 16 frame-tiles
    dim3 block(256);            // 8 wave32
    hipLaunchKernelGGL(bandsplit_kernel, grid, block, smem, stream,
                       spec, (float*)d_out, P);
}
